// SSDAttention_2920577762057
// MI455X (gfx1250) — matrix operand order repaired and checked by reading, not compiled, not run on hardware
//
#include <hip/hip_runtime.h>
#include <hip/hip_bf16.h>

// ---------------------------------------------------------------------------
// Problem constants (B=2, T=2048, E=1024, H=16, D=64)
// ---------------------------------------------------------------------------
#define BB 2
#define TT 2048
#define EE 1024
#define HH 16
#define DD 64
#define BT (BB * TT)          // 4096 tokens

typedef __attribute__((ext_vector_type(16))) __bf16 v16bf;
typedef __attribute__((ext_vector_type(8)))  __bf16 v8bf;
typedef __attribute__((ext_vector_type(8)))  float  v8f;
typedef __attribute__((ext_vector_type(4)))  unsigned int u32x4;
typedef __attribute__((ext_vector_type(4)))  int i32x4;
typedef __attribute__((ext_vector_type(8)))  int i32x8;

// Tensor Data Mover availability (device pass only)
#if defined(__HIP_DEVICE_COMPILE__) &&                         \
    __has_builtin(__builtin_amdgcn_tensor_load_to_lds) &&      \
    __has_builtin(__builtin_amdgcn_s_wait_tensorcnt)
#define USE_TDM 1
#else
#define USE_TDM 0
#endif

// ---------------------------------------------------------------------------
// WMMA helpers (CDNA5 wave32, v_wmma_f32_16x16x32_bf16)
// ---------------------------------------------------------------------------
static __device__ __forceinline__ v8f wmma_bf16(v16bf a, v16bf b, v8f c) {
  return __builtin_amdgcn_wmma_f32_16x16x32_bf16(
      /*neg_a=*/false, a, /*neg_b=*/false, b,
      /*c_mod=*/(short)0, c, /*reuse_a=*/false, /*reuse_b=*/false);
}

static __device__ __forceinline__ v16bf make16(v8bf lo, v8bf hi) {
  v16bf r;
#pragma unroll
  for (int i = 0; i < 8; ++i) { r[i] = lo[i]; r[i + 8] = hi[i]; }
  return r;
}

// A-matrix fragment (16xK tile slice, K=32): lanes 0-15 hold K {0..7,16..23},
// lanes 16-31 hold K {8..15,24..31} for row M = (passed-in row).
static __device__ __forceinline__ v16bf frag_a(const __bf16* tile, int row,
                                               int ld, int kbase, int lane) {
  const __bf16* p = tile + row * ld + kbase + ((lane >> 4) << 3);
  return make16(*(const v8bf*)p, *(const v8bf*)(p + 16));
}

// B-matrix fragment (32x16): lanes 0-15 hold K {0..7,16..23}, lanes 16-31
// hold K {8..15,24..31} for column N = lane&15. Tile stored N-major (row = N).
static __device__ __forceinline__ v16bf frag_b(const __bf16* tile, int nrow,
                                               int ld, int kbase, int lane) {
  const __bf16* p = tile + nrow * ld + kbase + ((lane >> 4) << 3);
  return make16(*(const v8bf*)p, *(const v8bf*)(p + 16));
}

// ---------------------------------------------------------------------------
// Tensor Data Mover: 2D tile DMA global -> LDS.
//  D# group0 (CDNA5 ISA): count=1, lds_addr, global_addr, type=2
//  D# group1 (8.4): data_size=2B, tensor dims (kept generously in-bounds),
//                   tile_dim0/1, tensor_dim0_stride.
//  LDS fill is linear (X fastest) -> packed [tile1][tile0] row-major, which is
//  exactly our LDS tile layout.  Issued once per wave (EXEC-independent),
//  completion via TENSORcnt.
// ---------------------------------------------------------------------------
static __device__ __forceinline__ unsigned lds_off(const void* p) {
  // LDS aperture: generic VA low 32 bits == workgroup LDS byte offset (ISA 10.2)
  return (unsigned)(uintptr_t)p;
}

#if USE_TDM
static __device__ __forceinline__ void tdm_load_2d(unsigned ldsoff,
                                                   const void* gaddr,
                                                   unsigned tile0_elems,
                                                   unsigned tile1_rows,
                                                   unsigned stride0_elems) {
  unsigned long long ga = (unsigned long long)(uintptr_t)gaddr;
  u32x4 g0;
  g0[0] = 1u;                                            // count=1 (valid D#)
  g0[1] = ldsoff;                                        // lds_addr
  g0[2] = (unsigned)ga;                                  // global_addr[31:0]
  g0[3] = (unsigned)((ga >> 32) & 0x01FFFFFFu) | (2u << 30);  // addr[56:32]|type=2
  const unsigned td0 = 1u << 20, td1 = 1u << 20;         // tile never OOB
  i32x8 g1;
  g1[0] = (int)(1u << 16);                               // data_size = 2 bytes
  g1[1] = (int)((td0 & 0xFFFFu) << 16);                  // tensor_dim0 lo
  g1[2] = (int)((td0 >> 16) | ((td1 & 0xFFFFu) << 16));  // dim0 hi | dim1 lo
  g1[3] = (int)((td1 >> 16) | (tile0_elems << 16));      // dim1 hi | tile_dim0
  g1[4] = (int)tile1_rows;                               // tile_dim1 (tile_dim2=0)
  g1[5] = (int)stride0_elems;                            // tensor_dim0_stride lo
  g1[6] = 0;
  g1[7] = 0;
  i32x4 z4 = {0, 0, 0, 0};
#if defined(__clang_major__) && __clang_major__ >= 23
  i32x8 z8 = {0, 0, 0, 0, 0, 0, 0, 0};
  __builtin_amdgcn_tensor_load_to_lds(g0, g1, z4, z4, z8, 0);
#else
  __builtin_amdgcn_tensor_load_to_lds(g0, g1, z4, z4, 0);
#endif
}
#endif

// ---------------------------------------------------------------------------
// Conversion / packing kernels
// ---------------------------------------------------------------------------
__global__ void cvt_f32_bf16(const float* __restrict__ in,
                             __bf16* __restrict__ out, int n) {
  int i = blockIdx.x * blockDim.x + threadIdx.x;
  if (i < n) out[i] = (__bf16)in[i];
}

// Pack Wq|Wk|Wv|Wg as transposed bf16: w4t[n][k] = W_sel[k][n&1023]
__global__ void pack_w4t(const float* __restrict__ Wq,
                         const float* __restrict__ Wk,
                         const float* __restrict__ Wv,
                         const float* __restrict__ Wg,
                         __bf16* __restrict__ out) {
  int idx = blockIdx.x * blockDim.x + threadIdx.x;   // 4M elements
  int n = idx >> 10, k = idx & 1023;
  int sel = n >> 10, col = n & 1023;
  const float* W = (sel == 0) ? Wq : (sel == 1) ? Wk : (sel == 2) ? Wv : Wg;
  out[(size_t)n * EE + k] = (__bf16)W[(size_t)k * EE + col];
}

__global__ void pack_wot(const float* __restrict__ Wo,
                         __bf16* __restrict__ out) {
  int idx = blockIdx.x * blockDim.x + threadIdx.x;   // 1M elements
  int n = idx >> 10, k = idx & 1023;
  out[(size_t)n * EE + k] = (__bf16)Wo[(size_t)k * EE + n];
}

// ---------------------------------------------------------------------------
// bf16 WMMA GEMM:  C[M,N] (f32) = A[M,K] (bf16, row-major) * Bt[N,K]^T (bf16)
// Block tile 128x128, 8 waves as 4(M) x 2(N), each wave 32x64.
// TDM double-buffered LDS staging: wave0 DMAs tile kt+1 while all compute kt.
// ---------------------------------------------------------------------------
__global__ __launch_bounds__(256) void gemm_bf16(
    const __bf16* __restrict__ A, const __bf16* __restrict__ Bt,
    float* __restrict__ C, int M, int N, int K) {
  __shared__ __bf16 As[2][128 * 32];
  __shared__ __bf16 Bs[2][128 * 32];

  const int tid  = threadIdx.x;
  const int lane = tid & 31;
  const int wave = tid >> 5;
  const int wm = wave >> 1;           // 0..3 -> 32-row slice
  const int wn = wave & 1;            // 0..1 -> 64-col slice
  const int m0 = blockIdx.y * 128;
  const int n0 = blockIdx.x * 128;
  const int nk = K / 32;

  v8f acc[2][4];
  const v8f vzero = {0.f, 0.f, 0.f, 0.f, 0.f, 0.f, 0.f, 0.f};
#pragma unroll
  for (int mi = 0; mi < 2; ++mi)
#pragma unroll
    for (int ni = 0; ni < 4; ++ni) acc[mi][ni] = vzero;

#if USE_TDM
  // prologue: DMA K-tile 0 into buffer 0
  if (wave == 0) {
    tdm_load_2d(lds_off(&As[0][0]), A + (size_t)m0 * K, 32, 128, (unsigned)K);
    tdm_load_2d(lds_off(&Bs[0][0]), Bt + (size_t)n0 * K, 32, 128, (unsigned)K);
    __builtin_amdgcn_s_wait_tensorcnt(0);
  }
  __syncthreads();

  for (int kt = 0; kt < nk; ++kt) {
    const int cur = kt & 1;
    if (wave == 0 && kt + 1 < nk) {   // async DMA of next K-tile (TENSORcnt)
      tdm_load_2d(lds_off(&As[cur ^ 1][0]),
                  A + (size_t)m0 * K + (kt + 1) * 32, 32, 128, (unsigned)K);
      tdm_load_2d(lds_off(&Bs[cur ^ 1][0]),
                  Bt + (size_t)n0 * K + (kt + 1) * 32, 32, 128, (unsigned)K);
    }
    if (wave == 1 && kt + 2 < nk) {   // L2 prefetch two tiles ahead
      __builtin_prefetch(A + (size_t)(m0 + lane * 4) * K + (kt + 2) * 32, 0, 1);
      __builtin_prefetch(Bt + (size_t)(n0 + lane * 4) * K + (kt + 2) * 32, 0, 1);
    }

    const __bf16* Ac = As[cur];
    const __bf16* Bc = Bs[cur];
    v16bf af[2];
#pragma unroll
    for (int mi = 0; mi < 2; ++mi)
      af[mi] = frag_a(Ac, wm * 32 + mi * 16 + (lane & 15), 32, 0, lane);
#pragma unroll
    for (int ni = 0; ni < 4; ++ni) {
      v16bf bf = frag_b(Bc, wn * 64 + ni * 16 + (lane & 15), 32, 0, lane);
#pragma unroll
      for (int mi = 0; mi < 2; ++mi)
        acc[mi][ni] = wmma_bf16(af[mi], bf, acc[mi][ni]);
    }

    __syncthreads();                                   // everyone done with cur
    if (wave == 0 && kt + 1 < nk)
      __builtin_amdgcn_s_wait_tensorcnt(0);            // next tile landed in LDS
    __syncthreads();
  }
#else
  // fallback: cooperative vector staging (16 contiguous bf16 per thread)
  const int e0 = tid * 16;
  const int sr = e0 >> 5;             // row in 128x32 tile
  const int sc = e0 & 31;             // col (0 or 16)
  for (int kt = 0; kt < nk; ++kt) {
    __syncthreads();
    {
      const uint4* gA = (const uint4*)(A + (size_t)(m0 + sr) * K + kt * 32 + sc);
      uint4* sA = (uint4*)(&As[0][0] + sr * 32 + sc);
      sA[0] = gA[0]; sA[1] = gA[1];
      const uint4* gB = (const uint4*)(Bt + (size_t)(n0 + sr) * K + kt * 32 + sc);
      uint4* sB = (uint4*)(&Bs[0][0] + sr * 32 + sc);
      sB[0] = gB[0]; sB[1] = gB[1];
      if (kt + 1 < nk) {
        __builtin_prefetch(A + (size_t)(m0 + sr) * K + (kt + 1) * 32 + sc, 0, 1);
        __builtin_prefetch(Bt + (size_t)(n0 + sr) * K + (kt + 1) * 32 + sc, 0, 1);
      }
    }
    __syncthreads();
    v16bf af[2];
#pragma unroll
    for (int mi = 0; mi < 2; ++mi)
      af[mi] = frag_a(&As[0][0], wm * 32 + mi * 16 + (lane & 15), 32, 0, lane);
#pragma unroll
    for (int ni = 0; ni < 4; ++ni) {
      v16bf bf = frag_b(&Bs[0][0], wn * 64 + ni * 16 + (lane & 15), 32, 0, lane);
#pragma unroll
      for (int mi = 0; mi < 2; ++mi)
        acc[mi][ni] = wmma_bf16(af[mi], bf, acc[mi][ni]);
    }
  }
#endif

  // C/D layout: element (M = r + 8*(lane>=16), N = lane&15)
  const int lh = lane >> 4, col = lane & 15;
#pragma unroll
  for (int mi = 0; mi < 2; ++mi)
#pragma unroll
    for (int ni = 0; ni < 4; ++ni)
#pragma unroll
      for (int r = 0; r < 8; ++r) {
        int row = m0 + wm * 32 + mi * 16 + r + 8 * lh;
        int cn  = n0 + wn * 64 + ni * 16 + col;
        C[(size_t)row * N + cn] = acc[mi][ni][r];
      }
}

// ---------------------------------------------------------------------------
// a-projection: la[b,h,t] = log(exp(-exp(A_log)*softplus(x@Wa + dt_bias)) + 1e-8)
// ---------------------------------------------------------------------------
__global__ void aproj_kernel(const float* __restrict__ x,
                             const float* __restrict__ Wa,
                             const float* __restrict__ A_log,
                             const float* __restrict__ dt_bias,
                             float* __restrict__ la) {
  __shared__ float xs[EE];
  const int bt = blockIdx.x;
  for (int e = threadIdx.x; e < EE; e += blockDim.x)
    xs[e] = x[(size_t)bt * EE + e];
  __syncthreads();
  if (threadIdx.x < HH) {
    const int h = threadIdx.x;
    float acc = dt_bias[h];
    for (int k = 0; k < EE; ++k) acc += xs[k] * Wa[(size_t)k * HH + h];
    float sp = (acc > 20.f) ? acc : log1pf(expf(acc));      // softplus
    float alpha = expf(-expf(A_log[h]) * sp);
    int b = bt >> 11, t = bt & (TT - 1);
    la[((size_t)(b * HH + h)) * TT + t] = logf(alpha + 1e-8f);
  }
}

// Inclusive cumsum over t per (b,h): 2048 elems, 256 threads x 8.
__global__ void scan_kernel(const float* __restrict__ la,
                            float* __restrict__ cl) {
  __shared__ float ssum[256];
  const int bh = blockIdx.x;
  const int tid = threadIdx.x;
  const float* in = la + (size_t)bh * TT;
  float* out = cl + (size_t)bh * TT;
  float v[8], s = 0.f;
#pragma unroll
  for (int j = 0; j < 8; ++j) { s += in[tid * 8 + j]; v[j] = s; }
  ssum[tid] = s;
  __syncthreads();
  for (int off = 1; off < 256; off <<= 1) {
    float t = (tid >= off) ? ssum[tid - off] : 0.f;
    __syncthreads();
    ssum[tid] += t;
    __syncthreads();
  }
  float excl = ssum[tid] - s;
#pragma unroll
  for (int j = 0; j < 8; ++j) out[tid * 8 + j] = v[j] + excl;
}

// ---------------------------------------------------------------------------
// L2-normalize q,k per head; write q/k/v as bf16 in [b,h,t,d] layout.
// qkvg layout: row bt, cols [0,1024)=q [1024,2048)=k [2048,3072)=v [3072,4096)=g
// ---------------------------------------------------------------------------
__global__ void qkvnorm_kernel(const float* __restrict__ qkvg,
                               __bf16* __restrict__ qn,
                               __bf16* __restrict__ kn,
                               __bf16* __restrict__ vn) {
  __shared__ float sq[HH], sk[HH];
  const int bt = blockIdx.x;
  const int tid = threadIdx.x;
  if (tid < HH) { sq[tid] = 0.f; sk[tid] = 0.f; }
  __syncthreads();
  const float* Q = qkvg + (size_t)bt * 4096;
  const float* K = Q + 1024;
  const float* V = Q + 2048;
  float qv[4], kv[4];
#pragma unroll
  for (int i = 0; i < 4; ++i) {
    int e = tid + i * 256;
    qv[i] = Q[e]; kv[i] = K[e];
    atomicAdd(&sq[e >> 6], qv[i] * qv[i]);
    atomicAdd(&sk[e >> 6], kv[i] * kv[i]);
  }
  __syncthreads();
  const int b = bt >> 11, t = bt & (TT - 1);
#pragma unroll
  for (int i = 0; i < 4; ++i) {
    int e = tid + i * 256;
    int h = e >> 6, d = e & 63;
    size_t o = ((size_t)(b * HH + h) * TT + t) * DD + d;
    float nq = fmaxf(sqrtf(sq[h]), 1e-12f);
    float nk2 = fmaxf(sqrtf(sk[h]), 1e-12f);
    qn[o] = (__bf16)(qv[i] / nq);
    kn[o] = (__bf16)(kv[i] / nk2);
    vn[o] = (__bf16)V[e];
  }
}

// ---------------------------------------------------------------------------
// Flash-style decayed attention per (b,h).  One block handles 64 query rows.
//   S = Q K^T (WMMA), S *= exp(cl_i - cl_j) causal, O += S V (WMMA),
//   RMSNorm over d + SiLU gate, store Y bf16 [bt, e].
// Q/K tiles staged via TDM (contiguous [64,64] bf16); V transposed manually.
// ---------------------------------------------------------------------------
__global__ __launch_bounds__(128) void attn_kernel(
    const __bf16* __restrict__ qn, const __bf16* __restrict__ kn,
    const __bf16* __restrict__ vn, const float* __restrict__ cl,
    const float* __restrict__ qkvg, const float* __restrict__ rms_w,
    __bf16* __restrict__ Y) {
  __shared__ __bf16 Qs[64 * 64];
  __shared__ __bf16 Ks[64 * 64];
  __shared__ __bf16 Vt[64 * 64];      // transposed: Vt[d][j]
  __shared__ __bf16 Sb[4][16 * 64];   // per-wave S tile (bf16 A operand)
  __shared__ float cli[64], clj[64];

  const int tid  = threadIdx.x;
  const int lane = tid & 31;
  const int wave = tid >> 5;
  const int lh   = lane >> 4;
  const int col  = lane & 15;
  const int bh   = blockIdx.y;               // b*H + h
  const int i0   = blockIdx.x * 64;
  const int b    = bh >> 4, h = bh & 15;

  // stage Q tile (rows i0..i0+63) + cl_i
#if USE_TDM
  if (wave == 0) {
    tdm_load_2d(lds_off(Qs), qn + ((size_t)bh * TT + i0) * DD, 64, 64, 64);
    __builtin_amdgcn_s_wait_tensorcnt(0);
  }
#else
  {
    const uint4* src = (const uint4*)(qn + ((size_t)bh * TT + i0) * DD);
    uint4* dst = (uint4*)Qs;
    for (int u = tid; u < 512; u += 128) dst[u] = src[u];
  }
#endif
  if (tid < 64) cli[tid] = cl[(size_t)bh * TT + i0 + tid];
  __syncthreads();

  // preload Q A-fragments (rows wave*16..+15, two 32-wide k chunks over d)
  v16bf qa[2];
#pragma unroll
  for (int kk = 0; kk < 2; ++kk)
    qa[kk] = frag_a(Qs, wave * 16 + col, 64, kk * 32, lane);

  v8f o[4];
  const v8f vzero = {0.f, 0.f, 0.f, 0.f, 0.f, 0.f, 0.f, 0.f};
#pragma unroll
  for (int ni = 0; ni < 4; ++ni) o[ni] = vzero;

  for (int jt = 0; jt <= blockIdx.x; ++jt) {
    const int j0 = jt * 64;
    __syncthreads();
    // stage K row-major (TDM), V transposed (manual), cl_j
#if USE_TDM
    if (wave == 0)
      tdm_load_2d(lds_off(Ks), kn + ((size_t)bh * TT + j0) * DD, 64, 64, 64);
#else
    {
      const uint4* src = (const uint4*)(kn + ((size_t)bh * TT + j0) * DD);
      uint4* dst = (uint4*)Ks;
      for (int u = tid; u < 512; u += 128) dst[u] = src[u];
    }
#endif
    {
      const __bf16* vsrc = vn + ((size_t)bh * TT + j0) * DD;
      for (int u = tid; u < 4096; u += 128) {
        int j = u >> 6, d = u & 63;
        Vt[d * 64 + j] = vsrc[u];
      }
      if (tid < 64) clj[tid] = cl[(size_t)bh * TT + j0 + tid];
    }
#if USE_TDM
    if (wave == 0) __builtin_amdgcn_s_wait_tensorcnt(0);
#endif
    __syncthreads();

    // S = Q K^T  (K-dim = d = 64 -> 2 wmma steps per 16x16 frag)
    v8f s[4];
#pragma unroll
    for (int ni = 0; ni < 4; ++ni) {
      s[ni] = vzero;
#pragma unroll
      for (int kk = 0; kk < 2; ++kk) {
        v16bf bf = frag_b(Ks, ni * 16 + col, 64, kk * 32, lane);
        s[ni] = wmma_bf16(qa[kk], bf, s[ni]);
      }
    }

    // decay + causal mask in f32 on C-layout, spill to LDS as bf16 A operand
#pragma unroll
    for (int ni = 0; ni < 4; ++ni)
#pragma unroll
      for (int r = 0; r < 8; ++r) {
        int li = wave * 16 + r + 8 * lh;        // local i within 64-row tile
        int lj = ni * 16 + col;                 // local j within 64-col tile
        int gi = i0 + li, gj = j0 + lj;
        float w = (gj <= gi) ? __expf(cli[li] - clj[lj]) : 0.f;
        Sb[wave][(r + 8 * lh) * 64 + lj] = (__bf16)(s[ni][r] * w);
      }

    // O += S V   (K-dim = j = 64 -> 2 wmma steps; B operand from Vt rows)
#pragma unroll
    for (int ni = 0; ni < 4; ++ni)              // d chunks of 16
#pragma unroll
      for (int jk = 0; jk < 2; ++jk) {
        v16bf af = frag_a(Sb[wave], col, 64, jk * 32, lane);
        v16bf bf = frag_b(Vt, ni * 16 + col, 64, jk * 32, lane);
        o[ni] = wmma_bf16(af, bf, o[ni]);
      }
  }

  // epilogue: RMSNorm over d (64) per row + SiLU gate, store bf16 Y[bt, e]
#pragma unroll
  for (int r = 0; r < 8; ++r) {
    float ss = 0.f;
#pragma unroll
    for (int ni = 0; ni < 4; ++ni) { float t = o[ni][r]; ss += t * t; }
    ss += __shfl_xor(ss, 1); ss += __shfl_xor(ss, 2);
    ss += __shfl_xor(ss, 4); ss += __shfl_xor(ss, 8);   // 16-lane half reduce
    float rinv = rsqrtf(ss * (1.f / 64.f) + 1e-6f);
    int gi = i0 + wave * 16 + r + 8 * lh;
    size_t bt = (size_t)b * TT + gi;
#pragma unroll
    for (int ni = 0; ni < 4; ++ni) {
      int d = ni * 16 + col;
      float g = qkvg[bt * 4096 + 3072 + h * 64 + d];
      float gate = g / (1.f + __expf(-g));              // silu
      float yv = o[ni][r] * rinv * rms_w[d] * gate;
      Y[bt * EE + h * 64 + d] = (__bf16)yv;
    }
  }
}

// ---------------------------------------------------------------------------
// Host-side launcher
// ---------------------------------------------------------------------------
extern "C" void kernel_launch(void* const* d_in, const int* in_sizes, int n_in,
                              void* d_out, int out_size, void* d_ws,
                              size_t ws_size, hipStream_t stream) {
  const float* x       = (const float*)d_in[0];
  const float* Wq      = (const float*)d_in[1];
  const float* Wk      = (const float*)d_in[2];
  const float* Wv      = (const float*)d_in[3];
  const float* Wa      = (const float*)d_in[4];
  const float* Wg      = (const float*)d_in[5];
  const float* Wo      = (const float*)d_in[6];
  const float* A_log   = (const float*)d_in[7];
  const float* dt_bias = (const float*)d_in[8];
  const float* rms_w   = (const float*)d_in[9];
  float* out = (float*)d_out;

  char* p = (char*)d_ws;
  __bf16* xb   = (__bf16*)(p + 0x00000000);   //  8 MiB  x bf16 [BT,E]
  __bf16* w4t  = (__bf16*)(p + 0x00800000);   //  8 MiB  Wq|Wk|Wv|Wg transposed
  __bf16* wot  = (__bf16*)(p + 0x01000000);   //  2 MiB  Wo transposed
  float*  qkvg = (float*) (p + 0x01200000);   // 64 MiB  [BT, 4E] f32
  float*  la   = (float*) (p + 0x05200000);   // 256 KiB log-alpha [bh, t]
  float*  cl   = (float*) (p + 0x05240000);   // 256 KiB cum-log   [bh, t]
  __bf16* qn   = (__bf16*)(p + 0x05280000);   //  8 MiB  [b,h,t,d]
  __bf16* kn   = (__bf16*)(p + 0x05A80000);   //  8 MiB
  __bf16* vn   = (__bf16*)(p + 0x06280000);   //  8 MiB
  __bf16* Yb   = (__bf16*)(p + 0x06A80000);   //  8 MiB  gated output bf16

  // 1) conversions / packing
  cvt_f32_bf16<<<(BT * EE) / 256, 256, 0, stream>>>(x, xb, BT * EE);
  pack_w4t<<<(4 * EE * EE) / 256, 256, 0, stream>>>(Wq, Wk, Wv, Wg, w4t);
  pack_wot<<<(EE * EE) / 256, 256, 0, stream>>>(Wo, wot);

  // 2) fused QKVG projection: [4096,4096] = xb[4096,1024] * w4t^T
  gemm_bf16<<<dim3(4 * EE / 128, BT / 128), 256, 0, stream>>>(
      xb, w4t, qkvg, BT, 4 * EE, EE);

  // 3) decay chain
  aproj_kernel<<<BT, 128, 0, stream>>>(x, Wa, A_log, dt_bias, la);
  scan_kernel<<<BB * HH, 256, 0, stream>>>(la, cl);

  // 4) head-wise normalize + relayout
  qkvnorm_kernel<<<BT, 256, 0, stream>>>(qkvg, qn, kn, vn);

  // 5) decayed attention + RMSNorm + gate
  attn_kernel<<<dim3(TT / 64, BB * HH), 128, 0, stream>>>(
      qn, kn, vn, cl, qkvg, rms_w, Yb);

  // 6) output projection -> f32 d_out
  gemm_bf16<<<dim3(EE / 128, BT / 128), 256, 0, stream>>>(
      Yb, wot, out, BT, EE, EE);
}
